// Net_58454504898772
// MI455X (gfx1250) — compile-verified, matrix operand order checked by reading, not run on hardware
//
#include <hip/hip_runtime.h>

// ---------------------------------------------------------------------------
// Types for WMMA
// ---------------------------------------------------------------------------
typedef __bf16 bf16x16 __attribute__((ext_vector_type(16)));
typedef float  f32x8   __attribute__((ext_vector_type(8)));
typedef unsigned int u32x4 __attribute__((ext_vector_type(4)));

union Frag {
  u32x4   q[2];
  bf16x16 v;
};

__device__ __forceinline__ unsigned short f2bf(float f) {
  unsigned u = __builtin_bit_cast(unsigned, f);
  u += 0x7FFFu + ((u >> 16) & 1u);   // round-to-nearest-even
  return (unsigned short)(u >> 16);
}

// ---------------------------------------------------------------------------
// Elementwise copy (agg = h before scatter-add)
// ---------------------------------------------------------------------------
__global__ void copy_f32_kernel(const float* __restrict__ s, float* __restrict__ d, long n) {
  long i = (long)blockIdx.x * blockDim.x + threadIdx.x;
  if (i < n) d[i] = s[i];
}

// ---------------------------------------------------------------------------
// Edge scatter-add: agg[dst] += h[src]. One wave (32 lanes) per edge.
// ---------------------------------------------------------------------------
__global__ void scatter_add_kernel(const float* __restrict__ h,
                                   const int* __restrict__ src,
                                   const int* __restrict__ dst,
                                   float* __restrict__ agg, int E, int K) {
  int e = blockIdx.x * 8 + (threadIdx.x >> 5);
  if (e >= E) return;
  int lane = threadIdx.x & 31;
  int s = src[e], d = dst[e];
  const float* hr = h + (size_t)s * K;
  float* ar = agg + (size_t)d * K;
  for (int c = lane; c < K; c += 32) atomicAdd(&ar[c], hr[c]);
}

// ---------------------------------------------------------------------------
// f32 [M x K] -> bf16 [M x Kpad], zero-padded
// ---------------------------------------------------------------------------
__global__ void pad_bf16_kernel(const float* __restrict__ src, unsigned short* __restrict__ dst,
                                int M, int K, int Kpad) {
  long i = (long)blockIdx.x * blockDim.x + threadIdx.x;
  if (i >= (long)M * Kpad) return;
  int m = (int)(i / Kpad), k = (int)(i % Kpad);
  dst[i] = (k < K) ? f2bf(src[(long)m * K + k]) : (unsigned short)0;
}

// ---------------------------------------------------------------------------
// Weight f32 [K x Nc] -> transposed bf16 Wt[Npad x Kpad], zero-padded
// ---------------------------------------------------------------------------
__global__ void wt_bf16_kernel(const float* __restrict__ w, unsigned short* __restrict__ wt,
                               int K, int Nc, int Kpad, int Npad) {
  long i = (long)blockIdx.x * blockDim.x + threadIdx.x;
  if (i >= (long)Npad * Kpad) return;
  int n = (int)(i / Kpad), k = (int)(i % Kpad);
  wt[i] = (k < K && n < Nc) ? f2bf(w[(long)k * Nc + n]) : (unsigned short)0;
}

// ---------------------------------------------------------------------------
// Segment sum + count (one wave per node row)
// ---------------------------------------------------------------------------
__global__ void seg_sum_kernel(const float* __restrict__ x, const int* __restrict__ idx,
                               float* __restrict__ gsum, float* __restrict__ gcnt,
                               int n, int K, int ldg, int coloff, int cslot) {
  int i = blockIdx.x * 8 + (threadIdx.x >> 5);
  if (i >= n) return;
  int lane = threadIdx.x & 31;
  int g = idx[i];
  const float* xr = x + (size_t)i * K;
  float* gr = gsum + (size_t)g * ldg + coloff;
  for (int c = lane; c < K; c += 32) atomicAdd(&gr[c], xr[c]);
  if (lane == 0) atomicAdd(&gcnt[g * 2 + cslot], 1.0f);
}

// Divide by counts and emit bf16 concat [G x 512] (cols 0-255 use cnt0, 256-511 cnt1)
__global__ void seg_fin_kernel(const float* __restrict__ gsum, const float* __restrict__ gcnt,
                               unsigned short* __restrict__ gcat, int g_total) {
  long i = (long)blockIdx.x * blockDim.x + threadIdx.x;
  if (i >= (long)g_total * 512) return;
  int g = (int)(i >> 9);
  int c = (int)(i & 511);
  float cnt = gcnt[g * 2 + (c >= 256 ? 1 : 0)];
  gcat[i] = f2bf(gsum[i] / fmaxf(cnt, 1.0f));
}

// ---------------------------------------------------------------------------
// WMMA bf16 GEMM: D[M x Ncols] = A[M x K] * Wt^T + bias, optional BN-affine +
// ReLU epilogue. Block = 4 waves, 64x64 output tile, B staged via LDS.
// A frag layout per ISA: lanes 0-15 M rows, q[0]=K[0..7], q[1]=K[16..23];
// lanes 16-31 same rows, q[0]=K[8..15], q[1]=K[24..31].
// ---------------------------------------------------------------------------
__global__ __launch_bounds__(128) void gemm_bf16_kernel(
    const unsigned short* __restrict__ A,     // [M x K] bf16 (K multiple of 32)
    const unsigned short* __restrict__ Bt,    // [Npad x K] bf16 (weights transposed)
    int M, int K,
    const float* __restrict__ bias, int Ncols,
    const float* __restrict__ bn_g, const float* __restrict__ bn_b,
    const float* __restrict__ bn_mean, const float* __restrict__ bn_var,
    int relu,
    float* __restrict__ outF, int ldf,
    unsigned short* __restrict__ outB, int ldb_out, int coloff)
{
  __shared__ __align__(16) unsigned short sB[64 * 40];  // 64 cols, stride 40 (80B) anti-conflict
  const int wave = threadIdx.x >> 5;
  const int lane = threadIdx.x & 31;
  const int ln   = lane & 15;
  const int hi   = lane >> 4;
  const int m0   = blockIdx.x * 64;
  const int n0   = blockIdx.y * 64;

  const int rowA = m0 + wave * 16 + ln;
  const unsigned short* Arow = A + (size_t)(rowA < M ? rowA : 0) * K;

  f32x8 acc[4] = {};

  for (int k0 = 0; k0 < K; k0 += 32) {
    // Stage Bt tile [64 x 32] into LDS (each thread: one 16B chunk)
    {
      const int r  = threadIdx.x >> 1;
      const int kc = (threadIdx.x & 1) * 16;
      *reinterpret_cast<u32x4*>(&sB[r * 40 + kc]) =
          *reinterpret_cast<const u32x4*>(Bt + (size_t)(n0 + r) * K + k0 + kc);
    }
    __syncthreads();

    Frag af;
    const int kb = k0 + hi * 8;
    af.q[0] = *reinterpret_cast<const u32x4*>(Arow + kb);
    af.q[1] = *reinterpret_cast<const u32x4*>(Arow + kb + 16);

#pragma unroll
    for (int j = 0; j < 4; ++j) {
      Frag bf;
      const unsigned short* bsrc = &sB[(j * 16 + ln) * 40 + hi * 8];
      bf.q[0] = *reinterpret_cast<const u32x4*>(bsrc);
      bf.q[1] = *reinterpret_cast<const u32x4*>(bsrc + 16);
      acc[j] = __builtin_amdgcn_wmma_f32_16x16x32_bf16(
          false, af.v, false, bf.v, (short)0, acc[j], false, false);
    }
    __syncthreads();
  }

  // Epilogue: bias + optional BN affine + optional ReLU; f32 and/or bf16 stores.
#pragma unroll
  for (int j = 0; j < 4; ++j) {
    const int col = n0 + j * 16 + ln;
    const bool cvalid = col < Ncols;
    float b = cvalid ? bias[col] : 0.f;
    float s = 1.f, t = 0.f;
    if (bn_g != nullptr && cvalid) {
      const float sc = bn_g[col] * rsqrtf(bn_var[col] + 1e-5f);
      t = bn_b[col] - bn_mean[col] * sc;
      s = sc;
    }
    f32x8 a = acc[j];
#pragma unroll
    for (int e = 0; e < 8; ++e) {
      const int r = m0 + wave * 16 + hi * 8 + e;
      if (r < M && cvalid) {
        float v = a[e] + b;
        v = v * s + t;
        if (relu) v = fmaxf(v, 0.f);
        if (outF) outF[(size_t)r * ldf + col] = v;
        if (outB) outB[(size_t)r * ldb_out + col + coloff] = f2bf(v);
      }
    }
  }
}

// ---------------------------------------------------------------------------
// Host orchestration
// ---------------------------------------------------------------------------
extern "C" void kernel_launch(void* const* d_in, const int* in_sizes, int n_in,
                              void* d_out, int out_size, void* d_ws, size_t ws_size,
                              hipStream_t stream) {
  (void)in_sizes; (void)n_in; (void)out_size; (void)ws_size;
  const int N = 50000, E = 800000, G = 2000, DIM = 256, FIN = 102, NCLS = 121;

  auto Pf = [&](int i) { return (const float*)d_in[i]; };
  auto Pi = [&](int i) { return (const int*)d_in[i]; };

  // Flat input indices (insertion-order recursive flatten of setup_inputs()).
  const int C11 = 5, C12 = 13, M1 = 21, C21 = 25, C22 = 33, M2 = 41,
            C31 = 45, C32 = 53, MG = 61, L1W = 65, L1B = 66, L2W = 67, L2B = 68;
  // gin param base pb: pb+0=l1.w pb+1=l1.b pb+2=l2.w pb+3=l2.b pb+4..7=bn g/b/mean/var

  // Workspace carve
  char* ws = (char*)d_ws;
  size_t off = 0;
  auto carve = [&](size_t bytes) -> char* {
    char* p = ws + off;
    off = (off + bytes + 255) & ~(size_t)255;
    return p;
  };
  float* h   = (float*)carve((size_t)N * DIM * 4);
  float* agg = (float*)carve((size_t)N * DIM * 4);
  unsigned short* aggb  = (unsigned short*)carve((size_t)N * DIM * 2);
  unsigned short* z1b   = (unsigned short*)carve((size_t)N * DIM * 2);
  unsigned short* catb  = (unsigned short*)carve((size_t)N * 2 * DIM * 2);
  unsigned short* wtb   = (unsigned short*)carve((size_t)4 << 20);
  float* gsum = (float*)carve((size_t)G * 2 * DIM * 4);
  float* gcnt = (float*)carve((size_t)G * 2 * 4);
  unsigned short* gcatb = (unsigned short*)carve((size_t)G * 2 * DIM * 2);
  unsigned short* gb1   = (unsigned short*)carve((size_t)G * DIM * 2);
  unsigned short* gb2   = (unsigned short*)carve((size_t)G * DIM * 2);

  // Convert all weights to transposed, padded bf16
  size_t wcur = 0;
  auto mkwt = [&](int pidx, int K, int Nc) -> unsigned short* {
    int Kpad = (K + 31) & ~31;
    int Npad = (Nc + 63) & ~63;
    unsigned short* p = wtb + wcur;
    wcur += (size_t)Kpad * Npad;
    long tot = (long)Kpad * Npad;
    wt_bf16_kernel<<<(tot + 255) / 256, 256, 0, stream>>>(Pf(pidx), p, K, Nc, Kpad, Npad);
    return p;
  };
  unsigned short* w_c11_1 = mkwt(C11 + 0, FIN, DIM);
  unsigned short* w_c11_2 = mkwt(C11 + 2, DIM, DIM);
  unsigned short* w_c12_1 = mkwt(C12 + 0, FIN, DIM);
  unsigned short* w_c12_2 = mkwt(C12 + 2, DIM, DIM);
  unsigned short* w_m1_1  = mkwt(M1 + 0, 2 * DIM, DIM);
  unsigned short* w_m1_2  = mkwt(M1 + 2, DIM, DIM);
  unsigned short* w_c21_1 = mkwt(C21 + 0, DIM, DIM);
  unsigned short* w_c21_2 = mkwt(C21 + 2, DIM, DIM);
  unsigned short* w_c22_1 = mkwt(C22 + 0, DIM, DIM);
  unsigned short* w_c22_2 = mkwt(C22 + 2, DIM, DIM);
  unsigned short* w_m2_1  = mkwt(M2 + 0, 2 * DIM, DIM);
  unsigned short* w_m2_2  = mkwt(M2 + 2, DIM, DIM);
  unsigned short* w_c31_1 = mkwt(C31 + 0, DIM, DIM);
  unsigned short* w_c31_2 = mkwt(C31 + 2, DIM, DIM);
  unsigned short* w_c32_1 = mkwt(C32 + 0, DIM, DIM);
  unsigned short* w_c32_2 = mkwt(C32 + 2, DIM, DIM);
  unsigned short* w_mg_1  = mkwt(MG + 0, 2 * DIM, DIM);
  unsigned short* w_mg_2  = mkwt(MG + 2, DIM, DIM);
  unsigned short* w_l1    = mkwt(L1W, DIM, DIM);
  unsigned short* w_l2    = mkwt(L2W, DIM, NCLS);

  auto gemm = [&](const unsigned short* Abf, int M, int K,
                  const unsigned short* Wt, int Npad, const float* bias, int Ncols,
                  const float* g, const float* bb, const float* mu, const float* var,
                  bool relu, float* oF, int ldf, unsigned short* oB, int ldb, int coff) {
    dim3 grid((M + 63) / 64, Npad / 64);
    gemm_bf16_kernel<<<grid, dim3(128), 0, stream>>>(Abf, Wt, M, K, bias, Ncols,
                                                     g, bb, mu, var, relu ? 1 : 0,
                                                     oF, ldf, oB, ldb, coff);
  };

  auto run_conv = [&](const float* hsrc, int Kin, const int* ei, int pb,
                      const unsigned short* wl1, const unsigned short* wl2, int coff) {
    int Kpad = (Kin + 31) & ~31;
    long tot = (long)N * Kin;
    copy_f32_kernel<<<(tot + 255) / 256, 256, 0, stream>>>(hsrc, agg, tot);
    scatter_add_kernel<<<(E + 7) / 8, 256, 0, stream>>>(hsrc, ei, ei + E, agg, E, Kin);
    long tp = (long)N * Kpad;
    pad_bf16_kernel<<<(tp + 255) / 256, 256, 0, stream>>>(agg, aggb, N, Kin, Kpad);
    // l1 + BN + ReLU -> z1b (bf16)
    gemm(aggb, N, Kpad, wl1, DIM, Pf(pb + 1), DIM,
         Pf(pb + 4), Pf(pb + 5), Pf(pb + 6), Pf(pb + 7), true,
         nullptr, 0, z1b, DIM, 0);
    // l2 + ReLU -> concat buffer (bf16) at column offset
    gemm(z1b, N, DIM, wl2, DIM, Pf(pb + 3), DIM,
         nullptr, nullptr, nullptr, nullptr, true,
         nullptr, 0, catb, 2 * DIM, coff);
  };

  auto run_mlp_nodes = [&](const unsigned short* wm1, const float* bm1,
                           const unsigned short* wm2, const float* bm2) {
    gemm(catb, N, 2 * DIM, wm1, DIM, bm1, DIM, nullptr, nullptr, nullptr, nullptr,
         true, nullptr, 0, z1b, DIM, 0);
    gemm(z1b, N, DIM, wm2, DIM, bm2, DIM, nullptr, nullptr, nullptr, nullptr,
         false, h, DIM, nullptr, 0, 0);
  };

  const float* x = Pf(0);
  const int* e1 = Pi(1);
  const int* e2 = Pi(2);
  const int* i1 = Pi(3);
  const int* i2 = Pi(4);

  // Layer 1 (input K = FIN)
  run_conv(x, FIN, e1, C11, w_c11_1, w_c11_2, 0);
  run_conv(x, FIN, e2, C12, w_c12_1, w_c12_2, DIM);
  run_mlp_nodes(w_m1_1, Pf(M1 + 1), w_m1_2, Pf(M1 + 3));
  // Layer 2
  run_conv(h, DIM, e1, C21, w_c21_1, w_c21_2, 0);
  run_conv(h, DIM, e2, C22, w_c22_1, w_c22_2, DIM);
  run_mlp_nodes(w_m2_1, Pf(M2 + 1), w_m2_2, Pf(M2 + 3));
  // Layer 3 (reference reuses mlp2)
  run_conv(h, DIM, e1, C31, w_c31_1, w_c31_2, 0);
  run_conv(h, DIM, e2, C32, w_c32_1, w_c32_2, DIM);
  run_mlp_nodes(w_m2_1, Pf(M2 + 1), w_m2_2, Pf(M2 + 3));

  // Segment means -> [G x 512] bf16 concat
  hipMemsetAsync(gsum, 0, (size_t)G * 2 * DIM * 4, stream);
  hipMemsetAsync(gcnt, 0, (size_t)G * 2 * 4, stream);
  seg_sum_kernel<<<(N + 7) / 8, 256, 0, stream>>>(h, i1, gsum, gcnt, N, DIM, 2 * DIM, 0, 0);
  seg_sum_kernel<<<(N + 7) / 8, 256, 0, stream>>>(h, i2, gsum, gcnt, N, DIM, 2 * DIM, DIM, 1);
  long gtot = (long)G * 2 * DIM;
  seg_fin_kernel<<<(gtot + 255) / 256, 256, 0, stream>>>(gsum, gcnt, gcatb, G);

  // Head: mlp -> lin1(ReLU) -> lin2 -> d_out [G x 121] f32
  gemm(gcatb, G, 2 * DIM, w_mg_1, DIM, Pf(MG + 1), DIM, nullptr, nullptr, nullptr, nullptr,
       true, nullptr, 0, gb1, DIM, 0);
  gemm(gb1, G, DIM, w_mg_2, DIM, Pf(MG + 3), DIM, nullptr, nullptr, nullptr, nullptr,
       false, nullptr, 0, gb2, DIM, 0);
  gemm(gb2, G, DIM, w_l1, DIM, Pf(L1B), DIM, nullptr, nullptr, nullptr, nullptr,
       true, nullptr, 0, gb1, DIM, 0);
  gemm(gb1, G, DIM, w_l2, 128, Pf(L2B), NCLS, nullptr, nullptr, nullptr, nullptr,
       false, (float*)d_out, NCLS, nullptr, 0, 0);
}
